// SpatialSelfAttention_Uncertain_28836410425444
// MI455X (gfx1250) — matrix operand order repaired and checked by reading, not compiled, not run on hardware
//
#include <hip/hip_runtime.h>
#include <hip/hip_bf16.h>

typedef _Float16 half_t;
typedef __attribute__((ext_vector_type(16))) _Float16 v16h;
typedef __attribute__((ext_vector_type(8)))  _Float16 v8h;
typedef __attribute__((ext_vector_type(8)))  float    v8f;
typedef unsigned int u32x4 __attribute__((ext_vector_type(4)));
typedef int          i32x8 __attribute__((ext_vector_type(8)));
typedef int          i32x4 __attribute__((ext_vector_type(4)));

#define HW 4096
#define CCH 256
#define NB 4
#define TS 264   // padded LDS row stride (halfs) for 256-wide tiles (+4 dwords / 128 dwords)
#define VS 72    // padded LDS row stride (halfs) for 64-wide tiles  (+4 dwords / 32 dwords)

#ifndef __has_builtin
#define __has_builtin(x) 0
#endif
#if __has_builtin(__builtin_amdgcn_tensor_load_to_lds) && __has_builtin(__builtin_amdgcn_s_wait_tensorcnt)
#define HAVE_TDM 1
#else
#define HAVE_TDM 0
#endif
#if defined(__has_include)
#if __has_include(<hip/amd_detail/amd_gfx1250_TDM.h>)
#define TDM_6ARG 1
#endif
#endif

union V16U { v16h v; v8h h[2]; };
union V8U  { v8h v; half_t e[8]; };

__device__ __forceinline__ v8f zero8() {
    v8f z = {0.f,0.f,0.f,0.f,0.f,0.f,0.f,0.f};
    return z;
}

__device__ __forceinline__ v8f wmma16(v16h a, v16h b, v8f c) {
    return __builtin_amdgcn_wmma_f32_16x16x32_f16(
        false, a, false, b, (short)0, c, false, false);
}

// A-matrix fragment (16x32 f16), interleaved K layout per ISA:
// lane (m = l&15, h = l>>4): K in [h*8, h*8+8) U [16+h*8, 16+h*8+8)
__device__ __forceinline__ v16h frag_a(const half_t* base, int ldm,
                                       int row0, int k0, int lane) {
    int m = lane & 15, hh = lane >> 4;
    const half_t* p = base + (size_t)(row0 + m) * ldm + k0 + hh * 8;
    V16U u;
    u.h[0] = *(const v8h*)p;
    u.h[1] = *(const v8h*)(p + 16);
    return u.v;
}

// B-matrix fragment (32x16 f16), column n per lane, interleaved K layout:
// lane (n = l&15, h = l>>4): K in [h*8, h*8+8) U [16+h*8, 16+h*8+8)
__device__ __forceinline__ v16h frag_b(const half_t* base, int ldm,
                                       int n0, int k0, int lane) {
    int n = lane & 15, hh = lane >> 4;
    const half_t* p = base + (size_t)(n0 + n) * ldm + k0 + hh * 8;
    V16U u;
    u.h[0] = *(const v8h*)p;
    u.h[1] = *(const v8h*)(p + 16);
    return u.v;
}

// ---------------- Tensor Data Mover: 2D strided tile -> padded LDS ----------------
// Builds D# per CDNA5 ISA and issues TENSOR_LOAD_TO_LDS.
// data_size=2B; pad codes: interval 0=2dw..7=256dw, amount 0=1dw..127=128dw.
__device__ __forceinline__ void tdm_load_2d(const half_t* gsrc, unsigned lds_off,
                                            unsigned tile_d0, unsigned tile_d1,
                                            unsigned stride_elems,
                                            unsigned pad_int_code, unsigned pad_amt_code) {
#if HAVE_TDM
    unsigned long long ga = (unsigned long long)(uintptr_t)gsrc;
    u32x4 g0 = {
        1u,                                            // count=1, user descriptor
        lds_off,                                       // lds_addr (bytes)
        (unsigned)(ga & 0xffffffffu),                  // global_addr[31:0]  (bits 95:64)
        (unsigned)((ga >> 32) & 0x01ffffffu) | (2u << 30) // addr[56:32], type=2
    };
    unsigned td0 = stride_elems;                       // tensor_dim0 >= tile_d0
    unsigned td1 = tile_d1;                            // tile fits exactly
    i32x8 g1 = {
        (int)((1u << 16) | (1u << 20) |                // data_size=2B, pad_enable
              (pad_int_code << 22) | (pad_amt_code << 25)),  // wg_mask=0
        (int)((td0 & 0xffffu) << 16),                  // abar_addr=0 | td0[15:0]
        (int)(((td0 >> 16) & 0xffffu) | ((td1 & 0xffffu) << 16)),
        (int)(((td1 >> 16) & 0xffffu) | ((tile_d0 & 0xffffu) << 16)),
        (int)(tile_d1 & 0xffffu),                      // tile_dim1 | tile_dim2=0
        (int)(stride_elems),                           // tensor_dim0_stride[31:0]
        0, 0                                           // stride hi, dim1_stride (2D)
    };
    i32x4 z4 = {0,0,0,0};
#ifdef TDM_6ARG
    i32x8 z8 = {0,0,0,0,0,0,0,0};
    __builtin_amdgcn_tensor_load_to_lds(g0, g1, z4, z4, z8, 0);
#else
    __builtin_amdgcn_tensor_load_to_lds(g0, g1, z4, z4, 0);
#endif
#endif
}

// ---------------- Stage 0: f32 -> f16 weight conversion ----------------
__global__ void cvt_kernel(const float* __restrict__ s, half_t* __restrict__ d, int n) {
    int i = blockIdx.x * 256 + threadIdx.x;
    if (i < n) d[i] = (half_t)s[i];
}

// ---------------- Stage 1: GroupNorm -> t [B][hw][C] f16 ----------------
__global__ __launch_bounds__(256) void gn_kernel(const float* __restrict__ x,
                                                 const float* __restrict__ gamma,
                                                 const float* __restrict__ beta,
                                                 half_t* __restrict__ t) {
    int bg = blockIdx.x;               // 0..127 = (b, g)
    int b = bg >> 5, g = bg & 31;
    const float* xp = x + ((size_t)b * CCH + g * 8) * HW;
    float s = 0.f, s2 = 0.f;
    for (int c = 0; c < 8; ++c)
        for (int i = threadIdx.x; i < HW; i += 256) {
            float v = xp[c * HW + i];
            s += v; s2 += v * v;
        }
    for (int m = 16; m >= 1; m >>= 1) {
        s  += __shfl_xor(s,  m);
        s2 += __shfl_xor(s2, m);
    }
    __shared__ float red[2][8];
    int wid = threadIdx.x >> 5, lane = threadIdx.x & 31;
    if (lane == 0) { red[0][wid] = s; red[1][wid] = s2; }
    __syncthreads();
    s = 0.f; s2 = 0.f;
    for (int w = 0; w < 8; ++w) { s += red[0][w]; s2 += red[1][w]; }
    float mu = s * (1.f / 32768.f);
    float var = s2 * (1.f / 32768.f) - mu * mu;
    float rinv = rsqrtf(var + 1e-6f);
    float gg[8], bb[8];
    for (int c = 0; c < 8; ++c) { gg[c] = gamma[g * 8 + c]; bb[c] = beta[g * 8 + c]; }
    for (int i = threadIdx.x; i < HW; i += 256) {
        V8U o;
        for (int c = 0; c < 8; ++c) {
            float v = (xp[c * HW + i] - mu) * rinv;
            o.e[c] = (half_t)(v * gg[c] + bb[c]);
        }
        *(v8h*)(t + ((size_t)b * HW + i) * CCH + g * 8) = o.v;
    }
}

// ---------------- Stage 2: Q,K,V,u projections (WMMA) ----------------
// MODE 0: dst row-major [i][C] at dst + ro*C + col
// MODE 1: dst transposed [C][hw] at dst + col*HW + ro
template <int MODE>
__device__ __forceinline__ void proj_mat(const half_t* __restrict__ W,
                                         const float* __restrict__ bias,
                                         const v16h* a, int lane,
                                         half_t* __restrict__ dst) {
    for (int nt = 0; nt < 16; ++nt) {
        v16h bf[8];
#pragma unroll
        for (int kc = 0; kc < 8; ++kc) bf[kc] = frag_b(W, CCH, nt * 16, kc * 32, lane);
        v8f acc = zero8();
#pragma unroll
        for (int kc = 0; kc < 8; ++kc) acc = wmma16(a[kc], bf[kc], acc);
        int col = nt * 16 + (lane & 15);
        float bb = bias[col];
#pragma unroll
        for (int r = 0; r < 8; ++r) {
            int ro = r + (lane >> 4) * 8;
            half_t hv = (half_t)(acc[r] + bb);
            if (MODE == 0) dst[(size_t)ro * CCH + col] = hv;
            else           dst[(size_t)col * HW + ro] = hv;
        }
    }
}

__global__ __launch_bounds__(128) void qkvu_kernel(
    const half_t* __restrict__ t,
    const half_t* __restrict__ wq16, const half_t* __restrict__ wk16,
    const half_t* __restrict__ wv16,
    const float* __restrict__ bq, const float* __restrict__ bk,
    const float* __restrict__ bv,
    const float* __restrict__ wu, const float* __restrict__ bu,
    half_t* __restrict__ qh, half_t* __restrict__ kh, half_t* __restrict__ vt,
    float* __restrict__ u, float* __restrict__ u_out) {
    int blk = blockIdx.x;              // 256 blocks of 64 tokens
    int b = blk >> 6;
    int i0 = (blk & 63) * 64;
    __shared__ __align__(16) half_t tl[64 * TS];
#if HAVE_TDM
    if ((threadIdx.x >> 5) == 0) {
        tdm_load_2d(t + (size_t)blk * 64 * CCH, (unsigned)(uintptr_t)tl,
                    CCH, 64, CCH, 6, 3);
        __builtin_amdgcn_s_wait_tensorcnt(0);
    }
#else
    {
        const uint4* src = (const uint4*)(t + (size_t)blk * 64 * CCH);
        for (int k = threadIdx.x; k < 2048; k += 128) {
            int row = k >> 5, seg = k & 31;
            ((uint4*)(tl + row * TS))[seg] = src[k];
        }
    }
#endif
    __syncthreads();

    int wave = threadIdx.x >> 5, lane = threadIdx.x & 31;
    v16h a[8];
#pragma unroll
    for (int kc = 0; kc < 8; ++kc) a[kc] = frag_a(tl, TS, wave * 16, kc * 32, lane);

    size_t rowRM = ((size_t)b * HW + i0 + wave * 16);
    proj_mat<0>(wq16, bq, a, lane, qh + rowRM * CCH);
    proj_mat<0>(wk16, bk, a, lane, kh + rowRM * CCH);
    proj_mat<1>(wv16, bv, a, lane, vt + (size_t)b * CCH * HW + (i0 + wave * 16));

    // u = sigmoid(t . wu + bu), one token per thread (f32 dot for accuracy)
    if (threadIdx.x < 64) {
        float sum = bu[0];
        const half_t* row = tl + threadIdx.x * TS;
        for (int c = 0; c < CCH; ++c) sum += (float)row[c] * wu[c];
        float uv = 1.f / (1.f + __expf(-sum));
        size_t gi = (size_t)b * HW + i0 + threadIdx.x;
        u[gi] = uv;
        u_out[gi] = uv;
    }
}

// ------- Stage 3: flash attention (WMMA + double-buffered TDM pipeline) -------
__global__ __launch_bounds__(128) void attn_kernel(
    const half_t* __restrict__ qh, const half_t* __restrict__ kh,
    const half_t* __restrict__ vt, const float* __restrict__ u,
    half_t* __restrict__ o) {
    int b = blockIdx.x >> 6;
    int q0 = (blockIdx.x & 63) * 64;
    __shared__ __align__(16) half_t Ql[64 * TS];
    __shared__ __align__(16) half_t Kl[2][64 * TS];
    __shared__ __align__(16) half_t Vl[2][CCH * VS];
    __shared__ __align__(16) half_t Pl[4][16 * VS];
    __shared__ float ul[2][64];
    int tid = threadIdx.x, wave = tid >> 5, lane = tid & 31;
    const float scale = 0.0625f;   // 256^-0.5

#if HAVE_TDM
    // Prologue: DMA Q tile and first K/V tiles (in-order on wave 0's TENSORcnt)
    if (wave == 0) {
        tdm_load_2d(qh + ((size_t)b * HW + q0) * CCH, (unsigned)(uintptr_t)Ql,
                    CCH, 64, CCH, 6, 3);
        tdm_load_2d(kh + (size_t)b * HW * CCH, (unsigned)(uintptr_t)Kl[0],
                    CCH, 64, CCH, 6, 3);
        tdm_load_2d(vt + (size_t)b * CCH * HW, (unsigned)(uintptr_t)Vl[0],
                    64, CCH, HW, 4, 3);
    }
    if (tid < 64) ul[0][tid] = u[(size_t)b * HW + tid] * scale;
#else
    {
        const uint4* src = (const uint4*)(qh + ((size_t)b * HW + q0) * CCH);
        for (int k = tid; k < 2048; k += 128) {
            int row = k >> 5, seg = k & 31;
            ((uint4*)(Ql + row * TS))[seg] = src[k];
        }
    }
#endif

    float mrow[8], lrow[8];
    for (int r = 0; r < 8; ++r) { mrow[r] = -1e30f; lrow[r] = 0.f; }
    v8f oacc[16];
    for (int ct = 0; ct < 16; ++ct) oacc[ct] = zero8();

    for (int jb = 0; jb < 64; ++jb) {
        int j0 = jb * 64, cur = jb & 1, nxt = cur ^ 1;
        __syncthreads();   // all waves done with buffer `nxt` (iteration jb-1)
#if HAVE_TDM
        if (jb + 1 < 64) {
            if (wave == 0) {   // DMA next tiles into the free buffer
                tdm_load_2d(kh + ((size_t)b * HW + j0 + 64) * CCH,
                            (unsigned)(uintptr_t)Kl[nxt], CCH, 64, CCH, 6, 3);
                tdm_load_2d(vt + (size_t)b * CCH * HW + (j0 + 64),
                            (unsigned)(uintptr_t)Vl[nxt], 64, CCH, HW, 4, 3);
            }
            if (tid < 64) ul[nxt][tid] = u[(size_t)b * HW + j0 + 64 + tid] * scale;
            if (tid < 32 && j0 + 128 < HW)   // two tiles ahead -> L2 (global_prefetch_b8)
                __builtin_prefetch(kh + ((size_t)b * HW + j0 + 128) * CCH + (size_t)tid * 512, 0, 1);
        }
        if (wave == 0) {
            // TDM completes in order: <=2 outstanding means current tiles are resident.
            if (jb + 1 < 64) __builtin_amdgcn_s_wait_tensorcnt(2);
            else             __builtin_amdgcn_s_wait_tensorcnt(0);
        }
#else
        {
            const uint4* ks = (const uint4*)(kh + ((size_t)b * HW + j0) * CCH);
            for (int k = tid; k < 2048; k += 128) {
                int row = k >> 5, seg = k & 31;
                ((uint4*)(Kl[cur] + row * TS))[seg] = ks[k];
            }
            for (int k = tid; k < 2048; k += 128) {
                int c = k >> 3, seg = k & 7;
                ((uint4*)(Vl[cur] + c * VS))[seg] =
                    *(const uint4*)(vt + ((size_t)b * CCH + c) * HW + j0 + seg * 8);
            }
        }
        if (tid < 64) ul[cur][tid] = u[(size_t)b * HW + j0 + tid] * scale;
#endif
        __syncthreads();
        const half_t* Kc = Kl[cur];
        const half_t* Vc = Vl[cur];
        const float*  uc = ul[cur];

        // S = Q K^T  (4 j-tiles of 16x16 per wave)
        v8f s[4];
        for (int jt = 0; jt < 4; ++jt) s[jt] = zero8();
#pragma unroll
        for (int kc = 0; kc < 8; ++kc) {
            v16h qa = frag_a(Ql, TS, wave * 16, kc * 32, lane);
            v16h bf[4];
#pragma unroll
            for (int jt = 0; jt < 4; ++jt) bf[jt] = frag_b(Kc, TS, jt * 16, kc * 32, lane);
#pragma unroll
            for (int jt = 0; jt < 4; ++jt) s[jt] = wmma16(qa, bf[jt], s[jt]);
        }
        // per-key uncertainty scale: s[i][j] *= scale * u[j]
        for (int jt = 0; jt < 4; ++jt) {
            float um = uc[jt * 16 + (lane & 15)];
            for (int r = 0; r < 8; ++r) s[jt][r] *= um;
        }
        // online softmax (rows live in 16-lane halves of the C layout)
        float corr[8];
#pragma unroll
        for (int r = 0; r < 8; ++r) {
            float mx = fmaxf(fmaxf(s[0][r], s[1][r]), fmaxf(s[2][r], s[3][r]));
            for (int m = 8; m >= 1; m >>= 1) mx = fmaxf(mx, __shfl_xor(mx, m));
            float mnew = fmaxf(mrow[r], mx);
            corr[r] = __expf(mrow[r] - mnew);
            mrow[r] = mnew;
            float rs = 0.f;
            for (int jt = 0; jt < 4; ++jt) {
                float p = __expf(s[jt][r] - mnew);
                s[jt][r] = p;
                rs += p;
            }
            for (int m = 8; m >= 1; m >>= 1) rs += __shfl_xor(rs, m);
            lrow[r] = lrow[r] * corr[r] + rs;
        }
        // C-layout f32 P -> A-layout f16 via per-wave LDS strip
        half_t* P = Pl[wave];
        for (int jt = 0; jt < 4; ++jt)
            for (int r = 0; r < 8; ++r) {
                int rr = r + (lane >> 4) * 8;
                P[rr * VS + jt * 16 + (lane & 15)] = (half_t)s[jt][r];
            }
        // rescale O accumulators
        for (int ct = 0; ct < 16; ++ct)
            for (int r = 0; r < 8; ++r) oacc[ct][r] *= corr[r];
        // O += P V   (same-wave LDS RAW; compiler inserts dscnt wait)
        for (int jc = 0; jc < 2; ++jc) {
            v16h pa = frag_a(P, VS, 0, jc * 32, lane);
            for (int cg = 0; cg < 4; ++cg) {
                v16h vb[4];
#pragma unroll
                for (int q = 0; q < 4; ++q)
                    vb[q] = frag_b(Vc, VS, (cg * 4 + q) * 16, jc * 32, lane);
#pragma unroll
                for (int q = 0; q < 4; ++q)
                    oacc[cg * 4 + q] = wmma16(pa, vb[q], oacc[cg * 4 + q]);
            }
        }
    }
    // finalize: O /= l, store f16 [b][i][c]
    for (int ct = 0; ct < 16; ++ct) {
        int col = ct * 16 + (lane & 15);
        for (int r = 0; r < 8; ++r) {
            int i = q0 + wave * 16 + r + (lane >> 4) * 8;
            o[((size_t)b * HW + i) * CCH + col] = (half_t)(oacc[ct][r] / lrow[r]);
        }
    }
}

// ---------------- Stage 4: out-proj + bias + residual (WMMA) ----------------
__global__ __launch_bounds__(128) void proj_kernel(
    const half_t* __restrict__ o, const half_t* __restrict__ wp16,
    const float* __restrict__ bp, const float* __restrict__ x,
    float* __restrict__ y) {
    int blk = blockIdx.x;
    int b = blk >> 6;
    int i0 = (blk & 63) * 64;
    __shared__ __align__(16) half_t ol[64 * TS];
    __shared__ float tb[4][16 * 17];
#if HAVE_TDM
    if ((threadIdx.x >> 5) == 0) {
        tdm_load_2d(o + ((size_t)b * HW + i0) * CCH, (unsigned)(uintptr_t)ol,
                    CCH, 64, CCH, 6, 3);
        __builtin_amdgcn_s_wait_tensorcnt(0);
    }
#else
    {
        const uint4* src = (const uint4*)(o + ((size_t)b * HW + i0) * CCH);
        for (int k = threadIdx.x; k < 2048; k += 128) {
            int row = k >> 5, seg = k & 31;
            ((uint4*)(ol + row * TS))[seg] = src[k];
        }
    }
#endif
    __syncthreads();
    int wave = threadIdx.x >> 5, lane = threadIdx.x & 31;
    v16h a[8];
#pragma unroll
    for (int kc = 0; kc < 8; ++kc) a[kc] = frag_a(ol, TS, wave * 16, kc * 32, lane);
    for (int ct = 0; ct < 16; ++ct) {
        v16h bf[8];
#pragma unroll
        for (int kc = 0; kc < 8; ++kc) bf[kc] = frag_b(wp16, CCH, ct * 16, kc * 32, lane);
        v8f acc = zero8();
#pragma unroll
        for (int kc = 0; kc < 8; ++kc) acc = wmma16(a[kc], bf[kc], acc);
        float* T = tb[wave];
#pragma unroll
        for (int r = 0; r < 8; ++r) {
            int rr = r + (lane >> 4) * 8;
            T[(lane & 15) * 17 + rr] = acc[r];   // transpose to [c][i]
        }
        // coalesced store along i with residual + bias (same-wave LDS RAW)
#pragma unroll
        for (int it = 0; it < 8; ++it) {
            int idx = lane + 32 * it;
            int cl = idx >> 4, il = idx & 15;
            int c = ct * 16 + cl;
            size_t gi = ((size_t)b * CCH + c) * HW + i0 + wave * 16 + il;
            y[gi] = x[gi] + bp[c] + T[cl * 17 + il];
        }
    }
}

extern "C" void kernel_launch(void* const* d_in, const int* in_sizes, int n_in,
                              void* d_out, int out_size, void* d_ws, size_t ws_size,
                              hipStream_t stream) {
    const float* x     = (const float*)d_in[0];
    const float* gamma = (const float*)d_in[1];
    const float* beta  = (const float*)d_in[2];
    const float* wq    = (const float*)d_in[3];
    const float* bq    = (const float*)d_in[4];
    const float* wk    = (const float*)d_in[5];
    const float* bk    = (const float*)d_in[6];
    const float* wv    = (const float*)d_in[7];
    const float* bv    = (const float*)d_in[8];
    const float* wu    = (const float*)d_in[9];
    const float* bu    = (const float*)d_in[10];
    const float* wp    = (const float*)d_in[11];
    const float* bp    = (const float*)d_in[12];

    char* w = (char*)d_ws;
    const size_t MB = 1024 * 1024;
    half_t* t16  = (half_t*)(w);                 // 8 MB  (reused as `o` after stage 2)
    half_t* qh   = (half_t*)(w + 8 * MB);        // 8 MB
    half_t* kh   = (half_t*)(w + 16 * MB);       // 8 MB
    half_t* vt   = (half_t*)(w + 24 * MB);       // 8 MB  (V transposed [B][C][hw])
    float*  uws  = (float*) (w + 32 * MB);       // 64 KB
    half_t* wq16 = (half_t*)(w + 32 * MB + 64 * 1024);
    half_t* wk16 = wq16 + 65536;
    half_t* wv16 = wk16 + 65536;
    half_t* wp16 = wv16 + 65536;

    float* yout = (float*)d_out;
    float* uout = yout + (size_t)NB * CCH * HW;

    cvt_kernel<<<256, 256, 0, stream>>>(wq, wq16, 65536);
    cvt_kernel<<<256, 256, 0, stream>>>(wk, wk16, 65536);
    cvt_kernel<<<256, 256, 0, stream>>>(wv, wv16, 65536);
    cvt_kernel<<<256, 256, 0, stream>>>(wp, wp16, 65536);

    gn_kernel<<<128, 256, 0, stream>>>(x, gamma, beta, t16);
    qkvu_kernel<<<256, 128, 0, stream>>>(t16, wq16, wk16, wv16,
                                         bq, bk, bv, wu, bu,
                                         qh, kh, vt, uws, uout);
    attn_kernel<<<256, 128, 0, stream>>>(qh, kh, vt, uws, t16 /* o reuses t */);
    proj_kernel<<<256, 128, 0, stream>>>(t16, wp16, bp, x, yout);
}
